// HSTULayer_84232898609278
// MI455X (gfx1250) — compile-verified, matrix operand order checked by reading, not run on hardware
//
#include <hip/hip_runtime.h>

// ---------------------------------------------------------------------------
// Types for CDNA5 WMMA (wave32): bf16 16x16x32, fp32 accumulate
// ---------------------------------------------------------------------------
typedef __attribute__((ext_vector_type(16))) __bf16 v16bf;
typedef __attribute__((ext_vector_type(8)))  __bf16 v8bf;
typedef __attribute__((ext_vector_type(8)))  float  v8f;
typedef __attribute__((ext_vector_type(4)))  int    v4i;

#define DEV __device__ __forceinline__

static constexpr int D    = 512;    // d_model
static constexpr int S    = 2048;   // seq len
static constexpr int Bsz  = 4;      // batch
static constexpr int H    = 8;      // heads
static constexpr int Dh   = 64;     // head dim
static constexpr int FF   = 2048;   // d_ff
static constexpr int M    = Bsz * S;        // 8192 rows
static constexpr int NQKV = 3 * D;          // 1536 fused qkv cols

static constexpr int LSTR = 72;     // padded LDS row stride (64 + 8 bf16 = +16B)

// Async global->LDS copies (CDNA5): guard on builtin availability so the file
// compiles on any toolchain; fallback is a plain load + ds_store pipeline.
#if defined(__has_builtin)
#if __has_builtin(__builtin_amdgcn_global_load_async_to_lds_b128) && \
    __has_builtin(__builtin_amdgcn_s_wait_asynccnt)
#define USE_ASYNC_LDS 1
#endif
#endif

#ifdef USE_ASYNC_LDS
#define WAIT_ASYNC(n) __builtin_amdgcn_s_wait_asynccnt(n)
#else
#define WAIT_ASYNC(n) ((void)0)
#endif

DEV __bf16 f2bf(float f) {
  unsigned u = __builtin_bit_cast(unsigned, f);
  unsigned r = (u + 0x7FFFu + ((u >> 16) & 1u)) >> 16;   // RNE
  unsigned short h = (unsigned short)r;
  return __builtin_bit_cast(__bf16, h);
}

struct BfPair { v8bf lo, hi; };

// Load one 16x32 bf16 A/B fragment (tile row stride `ld` elems).
// ISA layout (CDNA5 ISA): lane m = lane&15, hi-half holds K+8 / K+24 runs:
//   vector elems 0..7  <- K = hi*8 + 0..7
//   vector elems 8..15 <- K = 16 + hi*8 + 0..7
DEV v16bf load_frag(const __bf16* base, int ld, int lane) {
  int m  = lane & 15;
  int hi = lane >> 4;
  const __bf16* p = base + (size_t)m * ld + hi * 8;
  BfPair pr;
  pr.lo = *(const v8bf*)(p);
  pr.hi = *(const v8bf*)(p + 16);
  return __builtin_bit_cast(v16bf, pr);
}

DEV v8f wmma_bf16(v16bf a, v16bf b, v8f c) {
  return __builtin_amdgcn_wmma_f32_16x16x32_bf16(
      /*neg_a=*/false, a, /*neg_b=*/false, b,
      /*c_mod=*/(short)0, c, /*reuse_a=*/false, /*reuse_b=*/false);
}

// 16-byte global -> LDS copy (async on CDNA5, sync fallback otherwise)
DEV void copy16_g2l(const __bf16* g, __bf16* l) {
#ifdef USE_ASYNC_LDS
  __builtin_amdgcn_global_load_async_to_lds_b128(
      (__attribute__((address_space(1))) v4i*)(size_t)g,
      (__attribute__((address_space(3))) v4i*)(unsigned)(size_t)l,
      0, 0);
#else
  *(v8bf*)l = *(const v8bf*)g;
#endif
}

// ---------------------------------------------------------------------------
// LayerNorm (+optional residual add). One wave per row (512 elems, 16/lane).
// ---------------------------------------------------------------------------
__global__ __launch_bounds__(256) void ln_kernel(
    const float* __restrict__ x, const float* __restrict__ res,
    const float* __restrict__ g, const float* __restrict__ be,
    __bf16* __restrict__ xn, float* __restrict__ yout, int rows) {
  int wave = (blockIdx.x * blockDim.x + threadIdx.x) >> 5;
  int lane = threadIdx.x & 31;
  if (wave >= rows) return;
  const float* xr = x + (size_t)wave * D;
  float v[16];
  float sum = 0.f;
#pragma unroll
  for (int i = 0; i < 16; ++i) {
    int c = lane * 16 + i;
    float t = xr[c];
    if (res) t += res[(size_t)wave * D + c];
    v[i] = t; sum += t;
  }
#pragma unroll
  for (int m = 16; m >= 1; m >>= 1) sum += __shfl_xor(sum, m, 32);
  float mu = sum * (1.0f / D);
  float var = 0.f;
#pragma unroll
  for (int i = 0; i < 16; ++i) { float d = v[i] - mu; var += d * d; }
#pragma unroll
  for (int m = 16; m >= 1; m >>= 1) var += __shfl_xor(var, m, 32);
  float rs = rsqrtf(var * (1.0f / D) + 1e-5f);
#pragma unroll
  for (int i = 0; i < 16; ++i) {
    int c = lane * 16 + i;
    float o = (v[i] - mu) * rs * g[c] + be[c];
    xn[(size_t)wave * D + c] = f2bf(o);
    if (yout) yout[(size_t)wave * D + c] = v[i];
  }
}

// ---------------------------------------------------------------------------
// Weight convert+transpose: W [K x N] fp32  ->  Wt [N x K] bf16
// ---------------------------------------------------------------------------
__global__ __launch_bounds__(256) void wt_kernel(
    const float* __restrict__ W, __bf16* __restrict__ Wt, int K, int N) {
  int idx = blockIdx.x * 256 + threadIdx.x;
  if (idx >= K * N) return;
  int k = idx % K;
  int n = idx / K;
  Wt[(size_t)n * K + k] = f2bf(W[(size_t)k * N + n]);
}

// ---------------------------------------------------------------------------
// V transpose: qkv bf16 [M x 1536] (V at col 1024) -> Vt [B][H][Dh][S] bf16
// ---------------------------------------------------------------------------
__global__ __launch_bounds__(256) void vt_kernel(
    const __bf16* __restrict__ qkv, __bf16* __restrict__ Vt) {
  int idx = blockIdx.x * 256 + threadIdx.x;
  if (idx >= Bsz * S * D) return;
  int s = idx & (S - 1);
  int r = idx >> 11;          // b*512 + c
  int c = r & (D - 1);
  int b = r >> 9;
  int h  = c >> 6;
  int dh = c & 63;
  Vt[(((size_t)b * H + h) * Dh + dh) * S + s] =
      qkv[((size_t)b * S + s) * NQKV + 1024 + c];
}

// ---------------------------------------------------------------------------
// LDS-staged, async double-buffered WMMA GEMM block:
//   block = 256 thr (8 waves, 4x2), tile 128(M) x 128(N), K staged 64 at a time
//   As/Bs: [2 buf][128 rows][72 (padded 64)] bf16  ->  36 KB each
// ---------------------------------------------------------------------------
DEV void gemm_block(const __bf16* __restrict__ Ablk, const __bf16* __restrict__ Bblk,
                    int K, __bf16* As, __bf16* Bs, int tid, v8f acc[2][4]) {
  int lane = tid & 31, w = tid >> 5;
  int wm = w >> 1, wn = w & 1;

  auto stage = [&](int buf, int k0) {
#pragma unroll
    for (int i = 0; i < 4; ++i) {
      int c   = tid + i * 256;        // 0..1023 chunk id
      int row = c >> 3;               // 0..127
      int col = (c & 7) * 8;          // 0,8,..,56
      copy16_g2l(Ablk + (size_t)row * K + k0 + col,
                 As + (size_t)(buf * 128 + row) * LSTR + col);
      copy16_g2l(Bblk + (size_t)row * K + k0 + col,
                 Bs + (size_t)(buf * 128 + row) * LSTR + col);
    }
  };

  int nst = K / 64;
  stage(0, 0);
  for (int kt = 0; kt < nst; ++kt) {
    int buf = kt & 1;
    if (kt + 1 < nst) {
      stage(buf ^ 1, (kt + 1) * 64);
      WAIT_ASYNC(8);                  // 8 ops/stage outstanding allowed
    } else {
      WAIT_ASYNC(0);
    }
    __syncthreads();                  // stage `buf` visible to all waves
#pragma unroll
    for (int kc = 0; kc < 2; ++kc) {
      const __bf16* ap = As + (size_t)(buf * 128 + wm * 32) * LSTR + kc * 32;
      v16bf a0 = load_frag(ap, LSTR, lane);
      v16bf a1 = load_frag(ap + 16 * LSTR, LSTR, lane);
#pragma unroll
      for (int j = 0; j < 4; ++j) {
        v16bf b = load_frag(
            Bs + (size_t)(buf * 128 + wn * 64 + j * 16) * LSTR + kc * 32,
            LSTR, lane);
        acc[0][j] = wmma_bf16(a0, b, acc[0][j]);
        acc[1][j] = wmma_bf16(a1, b, acc[1][j]);
      }
    }
    __syncthreads();                  // done reading `buf` before re-fill
  }
}

// Fused QKV projection: xn[Mx512] * Wqkvt[1536x512]^T + bias -> qkv bf16 [Mx1536]
__global__ __launch_bounds__(256) void gemm_qkv(
    const __bf16* __restrict__ xn, const __bf16* __restrict__ Wt,
    const float* __restrict__ bq, const float* __restrict__ bk,
    const float* __restrict__ bv, __bf16* __restrict__ qkv) {
  __shared__ __bf16 As[2 * 128 * LSTR];
  __shared__ __bf16 Bs[2 * 128 * LSTR];
  int tid = threadIdx.x;
  int lane = tid & 31, w = tid >> 5;
  int row0 = blockIdx.x * 128 + (w >> 1) * 32;
  int col0 = blockIdx.y * 128 + (w & 1) * 64;
  v8f acc[2][4] = {};
  gemm_block(xn + (size_t)blockIdx.x * 128 * D,
             Wt + (size_t)blockIdx.y * 128 * D, D, As, Bs, tid, acc);
  int n = lane & 15, hi = lane >> 4;
#pragma unroll
  for (int i = 0; i < 2; ++i)
#pragma unroll
    for (int j = 0; j < 4; ++j) {
      int col = col0 + j * 16 + n;
      float bias = (col < 512) ? bq[col]
                 : (col < 1024) ? bk[col - 512] : bv[col - 1024];
#pragma unroll
      for (int r = 0; r < 8; ++r) {
        int row = row0 + i * 16 + r + hi * 8;
        qkv[(size_t)row * NQKV + col] = f2bf(acc[i][j][r] + bias);
      }
    }
}

// FFN1: xn2[Mx512] * W1t[2048x512]^T + b1, ReLU -> hbuf bf16 [Mx2048]
__global__ __launch_bounds__(256) void gemm_ffn1(
    const __bf16* __restrict__ xn2, const __bf16* __restrict__ W1t,
    const float* __restrict__ b1, __bf16* __restrict__ hbuf) {
  __shared__ __bf16 As[2 * 128 * LSTR];
  __shared__ __bf16 Bs[2 * 128 * LSTR];
  int tid = threadIdx.x;
  int lane = tid & 31, w = tid >> 5;
  int row0 = blockIdx.x * 128 + (w >> 1) * 32;
  int col0 = blockIdx.y * 128 + (w & 1) * 64;
  v8f acc[2][4] = {};
  gemm_block(xn2 + (size_t)blockIdx.x * 128 * D,
             W1t + (size_t)blockIdx.y * 128 * D, D, As, Bs, tid, acc);
  int n = lane & 15, hi = lane >> 4;
#pragma unroll
  for (int i = 0; i < 2; ++i)
#pragma unroll
    for (int j = 0; j < 4; ++j) {
      int col = col0 + j * 16 + n;
      float bias = b1[col];
#pragma unroll
      for (int r = 0; r < 8; ++r) {
        int row = row0 + i * 16 + r + hi * 8;
        float o = acc[i][j][r] + bias;
        hbuf[(size_t)row * FF + col] = f2bf(o > 0.f ? o : 0.f);
      }
    }
}

// FFN2: hbuf[Mx2048] * W2t[512x2048]^T + b2 + y -> out fp32 [Mx512]
__global__ __launch_bounds__(256) void gemm_ffn2(
    const __bf16* __restrict__ hbuf, const __bf16* __restrict__ W2t,
    const float* __restrict__ b2, const float* __restrict__ y,
    float* __restrict__ out) {
  __shared__ __bf16 As[2 * 128 * LSTR];
  __shared__ __bf16 Bs[2 * 128 * LSTR];
  int tid = threadIdx.x;
  int lane = tid & 31, w = tid >> 5;
  int row0 = blockIdx.x * 128 + (w >> 1) * 32;
  int col0 = blockIdx.y * 128 + (w & 1) * 64;
  v8f acc[2][4] = {};
  gemm_block(hbuf + (size_t)blockIdx.x * 128 * FF,
             W2t + (size_t)blockIdx.y * 128 * FF, FF, As, Bs, tid, acc);
  int n = lane & 15, hi = lane >> 4;
#pragma unroll
  for (int i = 0; i < 2; ++i)
#pragma unroll
    for (int j = 0; j < 4; ++j) {
      int col = col0 + j * 16 + n;
      float bias = b2[col];
#pragma unroll
      for (int r = 0; r < 8; ++r) {
        int row = row0 + i * 16 + r + hi * 8;
        out[(size_t)row * D + col] = acc[i][j][r] + bias + y[(size_t)row * D + col];
      }
    }
}

// ---------------------------------------------------------------------------
// Sigmoid attention. One wave owns 16 query rows of one (b,h); blocks cover
// 128 query rows. Scores via WMMA, sigmoid in VALU, attn tile re-fragmented
// through LDS as an A operand, out += attn * V via WMMA against Vt.
// ---------------------------------------------------------------------------
__global__ __launch_bounds__(256) void attn_kernel(
    const __bf16* __restrict__ qkv, const __bf16* __restrict__ Vt,
    float* __restrict__ attnout) {
  __shared__ __bf16 lds[8 * 16 * 64];     // 16 KB, one 16x64 tile per wave
  int lane = threadIdx.x & 31, w = threadIdx.x >> 5;
  int bh = blockIdx.y;
  int b = bh >> 3, h = bh & 7;
  int q0 = blockIdx.x * 128 + w * 16;

  const __bf16* qbase = qkv + ((size_t)b * S + q0) * NQKV + h * Dh;
  v16bf qa0 = load_frag(qbase,      NQKV, lane);   // K = 0..31 of Dh
  v16bf qa1 = load_frag(qbase + 32, NQKV, lane);   // K = 32..63

  v8f o[4] = {};
  const float rsc = 0.04419417382415922f;          // 1/sqrt(d_model=512)
  __bf16* my = lds + w * 16 * 64;
  int n = lane & 15, hi = lane >> 4;

  for (int t0 = 0; t0 < S; t0 += 64) {
    // scores[16 x 64] = Q (16xDh) * K^T (Dh x 64)
    v8f sc[4] = {};
#pragma unroll
    for (int j = 0; j < 4; ++j) {
      const __bf16* kb = qkv + ((size_t)b * S + t0 + j * 16) * NQKV + 512 + h * Dh;
      sc[j] = wmma_bf16(qa0, load_frag(kb,      NQKV, lane), sc[j]);
      sc[j] = wmma_bf16(qa1, load_frag(kb + 32, NQKV, lane), sc[j]);
    }
    // sigmoid, stash as bf16 in LDS (C-layout -> memory row-major)
#pragma unroll
    for (int j = 0; j < 4; ++j)
#pragma unroll
      for (int r = 0; r < 8; ++r) {
        float sv = 1.0f / (1.0f + __expf(-sc[j][r] * rsc));
        my[(r + hi * 8) * 64 + j * 16 + n] = f2bf(sv);
      }
    __syncthreads();
    // out[16 x Dh] += attn (16 x 64 keys) * V (64 keys x Dh)
    v16bf aa0 = load_frag(my,      64, lane);
    v16bf aa1 = load_frag(my + 32, 64, lane);
#pragma unroll
    for (int j = 0; j < 4; ++j) {
      const __bf16* vb = Vt + (((size_t)b * H + h) * Dh + j * 16) * S + t0;
      o[j] = wmma_bf16(aa0, load_frag(vb,      S, lane), o[j]);
      o[j] = wmma_bf16(aa1, load_frag(vb + 32, S, lane), o[j]);
    }
  }

  float* obase = attnout + ((size_t)b * S + q0) * D + h * Dh;
#pragma unroll
  for (int j = 0; j < 4; ++j)
#pragma unroll
    for (int r = 0; r < 8; ++r)
      obase[(size_t)(r + hi * 8) * D + j * 16 + n] = o[j][r];
}

// ---------------------------------------------------------------------------
// Host orchestration
// ---------------------------------------------------------------------------
extern "C" void kernel_launch(void* const* d_in, const int* in_sizes, int n_in,
                              void* d_out, int out_size, void* d_ws, size_t ws_size,
                              hipStream_t stream) {
  const float* x   = (const float*)d_in[0];
  const float* Wq  = (const float*)d_in[1];
  const float* bq  = (const float*)d_in[2];
  const float* Wk  = (const float*)d_in[3];
  const float* bk  = (const float*)d_in[4];
  const float* Wv  = (const float*)d_in[5];
  const float* bv  = (const float*)d_in[6];
  const float* W1  = (const float*)d_in[7];
  const float* b1  = (const float*)d_in[8];
  const float* W2  = (const float*)d_in[9];
  const float* b2  = (const float*)d_in[10];
  const float* g1  = (const float*)d_in[11];
  const float* be1 = (const float*)d_in[12];
  const float* g2  = (const float*)d_in[13];
  const float* be2 = (const float*)d_in[14];
  float* out = (float*)d_out;

  char* ws = (char*)d_ws;
  auto alloc = [&](size_t bytes) -> void* {
    void* p = (void*)ws;
    ws += (bytes + 255) & ~(size_t)255;
    return p;
  };
  __bf16* xn      = (__bf16*)alloc((size_t)M * D * 2);
  __bf16* Wqkvt   = (__bf16*)alloc((size_t)NQKV * D * 2);
  __bf16* qkv     = (__bf16*)alloc((size_t)M * NQKV * 2);
  __bf16* Vt      = (__bf16*)alloc((size_t)Bsz * H * Dh * S * 2);
  float*  attnout = (float*) alloc((size_t)M * D * 4);
  float*  y       = (float*) alloc((size_t)M * D * 4);
  __bf16* xn2     = (__bf16*)alloc((size_t)M * D * 2);
  __bf16* W1t     = (__bf16*)alloc((size_t)FF * D * 2);
  __bf16* W2t     = (__bf16*)alloc((size_t)D * FF * 2);
  __bf16* hbuf    = (__bf16*)alloc((size_t)M * FF * 2);

  // 1) LN1 -> xn (bf16)
  ln_kernel<<<M / 8, 256, 0, stream>>>(x, nullptr, g1, be1, xn, nullptr, M);

  // 2) weight transpose+convert
  wt_kernel<<<(D * D + 255) / 256, 256, 0, stream>>>(Wq, Wqkvt,            D, D);
  wt_kernel<<<(D * D + 255) / 256, 256, 0, stream>>>(Wk, Wqkvt + 512 * D,  D, D);
  wt_kernel<<<(D * D + 255) / 256, 256, 0, stream>>>(Wv, Wqkvt + 1024 * D, D, D);
  wt_kernel<<<(D * FF + 255) / 256, 256, 0, stream>>>(W1, W1t, D,  FF);
  wt_kernel<<<(FF * D + 255) / 256, 256, 0, stream>>>(W2, W2t, FF, D);

  // 3) fused QKV projection (WMMA, async LDS staging)
  gemm_qkv<<<dim3(M / 128, NQKV / 128), 256, 0, stream>>>(xn, Wqkvt, bq, bk, bv, qkv);

  // 4) V transpose for attn*V B-operands
  vt_kernel<<<(Bsz * S * D + 255) / 256, 256, 0, stream>>>(qkv, Vt);

  // 5) sigmoid attention (WMMA)
  attn_kernel<<<dim3(S / 128, Bsz * H), 256, 0, stream>>>(qkv, Vt, attnout);

  // 6) residual + LN2 -> xn2 (bf16), y (fp32 residual)
  ln_kernel<<<M / 8, 256, 0, stream>>>(x, attnout, g2, be2, xn2, y, M);

  // 7) FFN (WMMA, async LDS staging)
  gemm_ffn1<<<dim3(M / 128, FF / 128), 256, 0, stream>>>(xn2, W1t, b1, hbuf);
  gemm_ffn2<<<dim3(M / 128, D / 128), 256, 0, stream>>>(hbuf, W2t, b2, y, out);
}
